// MyOURS_empty_7713761263777
// MI455X (gfx1250) — matrix operand order repaired and checked by reading, not compiled, not run on hardware
//
#include <hip/hip_runtime.h>
#include <hip/hip_bf16.h>
#include <cstdint>
#include <cstddef>

// ---------------------------------------------------------------------------
// GMAN-ish STE + DCGRU (support == identity -> plain GRU with folded weights)
// B=64, P=Q=12, N=1024, D=64.  Main GEMMs run on v_wmma_f32_16x16x32_bf16.
// f32->bf16 conversions use the native __bf16 cast so the backend can select
// v_cvt_pk_bf16_f32 instead of a manual bit-twiddled RNE sequence.
// ---------------------------------------------------------------------------

typedef __attribute__((ext_vector_type(16))) __bf16 v16bf;
typedef __attribute__((ext_vector_type(8)))  float  v8f;

union BFrag {
  v16bf v;
  __bf16 h[16];
  unsigned short s[16];
  uint4 q[2];
};

__device__ __forceinline__ unsigned short f2bf(float f) {
  __bf16 h = (__bf16)f;                      // fptrunc -> v_cvt_pk_bf16_f32
  return __builtin_bit_cast(unsigned short, h);
}

__device__ __forceinline__ float sigmoidf_(float x) {
  return 1.0f / (1.0f + __expf(-x));
}

// A-operand (16x32 bf16 tile) from a bf16 matrix [rows x ld].
// ISA layout: lane m=lane&15, group g=lane>>4; s[0..7] <-> K = k0+g*8+{0..7},
// s[8..15] <-> K = k0+16+g*8+{0..7}.
__device__ __forceinline__ v16bf loadA_bf16(const unsigned short* base, int rowBase,
                                            int ld, int k0, int lane) {
  const int m = lane & 15, g = lane >> 4;
  const unsigned short* p = base + (size_t)(rowBase + m) * ld + k0 + g * 8;
  BFrag f;
  f.q[0] = *(const uint4*)(p);
  f.q[1] = *(const uint4*)(p + 16);
  return f.v;
}

// Same A layout but converting from f32 source on the fly.
__device__ __forceinline__ v16bf loadA_f32(const float* base, int rowBase,
                                           int ld, int k0, int lane) {
  const int m = lane & 15, g = lane >> 4;
  const float* p = base + (size_t)(rowBase + m) * ld + k0 + g * 8;
  BFrag f;
#pragma unroll
  for (int e = 0; e < 8; ++e) f.h[e]     = (__bf16)p[e];
#pragma unroll
  for (int e = 0; e < 8; ++e) f.h[8 + e] = (__bf16)p[16 + e];
  return f.v;
}

// B-operand (32x16 bf16 tile) from pre-transposed weights WT [outDim x K]:
// lane n=lane&15 holds column colBase+n; s[0..7] <-> K = k0+g*8+{0..7}, s[8..15] <-> K = k0+16+g*8+{0..7}.
__device__ __forceinline__ v16bf loadB_w(const unsigned short* WT, int colBase,
                                         int ld, int k0, int lane) {
  const int n = lane & 15, g = lane >> 4;
  const unsigned short* p = WT + (size_t)(colBase + n) * ld + k0 + g * 8;
  BFrag f;
  f.q[0] = *(const uint4*)(p);
  f.q[1] = *(const uint4*)(p + 16);
  return f.v;
}

__device__ __forceinline__ v8f wmma_bf16(v16bf a, v16bf b, v8f c) {
  return __builtin_amdgcn_wmma_f32_16x16x32_bf16(false, a, false, b, (short)0, c,
                                                 false, false);
}

// ---------------------------------------------------------------------------
// Weight preparation
// ---------------------------------------------------------------------------

// Fold gconv weight W [256 x outDim] (z=[x,s], z1=z) into bf16 transposed
// WxT/WsT [outDim x 64].
__global__ void k_fold(const float* __restrict__ W, unsigned short* __restrict__ WxT,
                       unsigned short* __restrict__ WsT, int outDim) {
  const int i = blockIdx.x * blockDim.x + threadIdx.x;
  if (i >= outDim * 64) return;
  const int j = i >> 6, k = i & 63;
  WxT[i] = f2bf(W[(size_t)k * outDim + j]        + W[(size_t)(128 + k) * outDim + j]);
  WsT[i] = f2bf(W[(size_t)(64 + k) * outDim + j] + W[(size_t)(192 + k) * outDim + j]);
}

// Transpose f32 W [64 x Nc] into bf16 WT [NcPad x 64] (zero-padded rows).
__global__ void k_transpose(const float* __restrict__ W, unsigned short* __restrict__ WT,
                            int Nc, int NcPad) {
  const int i = blockIdx.x * blockDim.x + threadIdx.x;
  if (i >= NcPad * 64) return;
  const int j = i >> 6, k = i & 63;
  WT[i] = (j < Nc) ? f2bf(W[(size_t)k * Nc + j]) : (unsigned short)0;
}

// ---------------------------------------------------------------------------
// Small scalar helpers (STEmbedding path; tiny FLOPs)
// ---------------------------------------------------------------------------

__global__ void k_zero(float* __restrict__ p, int n) {
  const int i = blockIdx.x * blockDim.x + threadIdx.x;
  if (i < n) p[i] = 0.0f;
}

__global__ void k_mm_small(const float* __restrict__ A, const float* __restrict__ W,
                           const float* __restrict__ bias, float* __restrict__ out,
                           int M, int K, int Nc, int doRelu) {
  const int i = blockIdx.x * blockDim.x + threadIdx.x;
  if (i >= M * Nc) return;
  const int r = i / Nc, c = i % Nc;
  float s = bias[c];
  for (int k = 0; k < K; ++k) s += A[(size_t)r * K + k] * W[(size_t)k * Nc + c];
  out[i] = doRelu ? fmaxf(s, 0.0f) : s;
}

// te_oh @ W_te1 is a two-row gather of W_te1 (one-hot).
__global__ void k_te1(const int* __restrict__ TE, const float* __restrict__ Wte1,
                      const float* __restrict__ bte1, float* __restrict__ out) {
  const int i = blockIdx.x * blockDim.x + threadIdx.x;
  if (i >= 64 * 24 * 64) return;
  const int row = i >> 6, j = i & 63;
  const int d0 = TE[row * 2 + 0];
  const int d1 = TE[row * 2 + 1];
  const float v = Wte1[(size_t)d0 * 64 + j] + Wte1[(size_t)(7 + d1) * 64 + j] + bte1[j];
  out[i] = fmaxf(v, 0.0f);
}

// ---------------------------------------------------------------------------
// FC_X_in + STE add  ->  Xe (bf16, time-major [P][B][N][64])
// grid = (P*B*N)/16 blocks, 128 threads (4 waves = 4 col tiles). 2 WMMAs/wave.
// ---------------------------------------------------------------------------
__global__ void __launch_bounds__(128) k_fc_in(
    const float* __restrict__ X,            // [B][12][1024]
    const float* __restrict__ w1,           // [64]  (W_in1 row)
    const float* __restrict__ b1,           // [64]
    const unsigned short* __restrict__ W2T, // bf16 [64][64] transposed W_in2
    const float* __restrict__ b2,           // [64]
    const float* __restrict__ se,           // [1024][64]
    const float* __restrict__ te,           // [B*24][64]
    unsigned short* __restrict__ Xe)        // bf16 [P*B*N][64]
{
  const int lane = threadIdx.x & 31;
  const int w    = threadIdx.x >> 5;
  const int rowBase = blockIdx.x << 4;
  const int colBase = w << 4;
  const int m = lane & 15, g = lane >> 4;

  const int row0 = rowBase + m;           // time-major row id
  const int p  = row0 >> 16;              // B*N = 65536
  const int bq = (row0 >> 10) & 63;
  const int n0 = row0 & 1023;
  const float x = X[((size_t)(bq * 12 + p) << 10) + n0];

  v8f acc = {};
#pragma unroll
  for (int kc = 0; kc < 64; kc += 32) {
    BFrag fa;
    const float* wl = w1 + kc + g * 8;
    const float* bl = b1 + kc + g * 8;
#pragma unroll
    for (int e = 0; e < 8; ++e) fa.h[e]     = (__bf16)fmaxf(0.0f, x * wl[e] + bl[e]);
#pragma unroll
    for (int e = 0; e < 8; ++e) fa.h[8 + e] = (__bf16)fmaxf(0.0f, x * wl[16 + e] + bl[16 + e]);
    const v16bf fb = loadB_w(W2T, colBase, 64, kc, lane);
    acc = wmma_bf16(fa.v, fb, acc);
  }

  const int d = colBase + (lane & 15);
  const float bias = b2[d] + te[(size_t)(bq * 24 + p) * 64 + d];
#pragma unroll
  for (int v = 0; v < 8; ++v) {
    const int row = rowBase + v + (g << 3);
    const int n   = row & 1023;
    Xe[(size_t)row * 64 + d] = f2bf(acc[v] + bias + se[(size_t)n * 64 + d]);
  }
}

// ---------------------------------------------------------------------------
// GRU gate kernel:  ru = sigmoid(x@Wgx + s@Wgs + bg); emit RS = r*s (bf16), U.
// grid = M/16 blocks, 256 threads (8 waves = 8 col tiles of 128). 4 WMMAs/wave.
// ---------------------------------------------------------------------------
__global__ void __launch_bounds__(256) k_gru_gates(
    const unsigned short* __restrict__ Xt,   // bf16 [M][64]
    const float* __restrict__ S,             // f32  [M][64]
    const unsigned short* __restrict__ WgxT, // bf16 [128][64]
    const unsigned short* __restrict__ WgsT, // bf16 [128][64]
    const float* __restrict__ bg,            // [128]
    unsigned short* __restrict__ RS,         // bf16 [M][64]
    float* __restrict__ U)                   // f32  [M][64]
{
  const int lane = threadIdx.x & 31;
  const int w    = threadIdx.x >> 5;
  const int rowBase = blockIdx.x << 4;
  const int colBase = w << 4;

  v8f acc = {};
#pragma unroll
  for (int kc = 0; kc < 64; kc += 32) {
    const v16bf a = loadA_bf16(Xt, rowBase, 64, kc, lane);
    const v16bf bb = loadB_w(WgxT, colBase, 64, kc, lane);
    acc = wmma_bf16(a, bb, acc);
  }
#pragma unroll
  for (int kc = 0; kc < 64; kc += 32) {
    const v16bf a = loadA_f32(S, rowBase, 64, kc, lane);
    const v16bf bb = loadB_w(WgsT, colBase, 64, kc, lane);
    acc = wmma_bf16(a, bb, acc);
  }

  const int g = lane >> 4;
  const int col = colBase + (lane & 15);
  const float bias = bg[col];
#pragma unroll
  for (int v = 0; v < 8; ++v) {
    const int row = rowBase + v + (g << 3);
    const float val = sigmoidf_(acc[v] + bias);
    if (w < 4) {                                  // r-half -> RS = r * state
      RS[(size_t)row * 64 + col] = f2bf(val * S[(size_t)row * 64 + col]);
    } else {                                      // u-half
      U[(size_t)row * 64 + (col - 64)] = val;
    }
  }
}

// ---------------------------------------------------------------------------
// GRU candidate + state update:
//   c = tanh(x@Wcx + RS@Wcs + bc);  new = u*s + (1-u)*c
// grid = M/16 blocks, 128 threads (4 waves). 4 WMMAs/wave.
// ---------------------------------------------------------------------------
__global__ void __launch_bounds__(128) k_gru_cand(
    const unsigned short* __restrict__ Xt,   // bf16 [M][64]
    const unsigned short* __restrict__ RS,   // bf16 [M][64]
    const unsigned short* __restrict__ WcxT, // bf16 [64][64]
    const unsigned short* __restrict__ WcsT, // bf16 [64][64]
    const float* __restrict__ bc,            // [64]
    const float* __restrict__ U,             // [M][64]
    const float* __restrict__ S,             // [M][64]
    float* __restrict__ Snew,                // [M][64]
    unsigned short* __restrict__ seqOut)     // bf16 [M][64] or nullptr
{
  const int lane = threadIdx.x & 31;
  const int w    = threadIdx.x >> 5;
  const int rowBase = blockIdx.x << 4;
  const int colBase = w << 4;

  v8f acc = {};
#pragma unroll
  for (int kc = 0; kc < 64; kc += 32) {
    const v16bf a = loadA_bf16(Xt, rowBase, 64, kc, lane);
    const v16bf bb = loadB_w(WcxT, colBase, 64, kc, lane);
    acc = wmma_bf16(a, bb, acc);
  }
#pragma unroll
  for (int kc = 0; kc < 64; kc += 32) {
    const v16bf a = loadA_bf16(RS, rowBase, 64, kc, lane);
    const v16bf bb = loadB_w(WcsT, colBase, 64, kc, lane);
    acc = wmma_bf16(a, bb, acc);
  }

  const int g = lane >> 4;
  const int col = colBase + (lane & 15);
  const float bias = bc[col];
#pragma unroll
  for (int v = 0; v < 8; ++v) {
    const int row = rowBase + v + (g << 3);
    const size_t idx = (size_t)row * 64 + col;
    const float c = tanhf(acc[v] + bias);
    const float u = U[idx];
    const float o = u * S[idx] + (1.0f - u) * c;
    Snew[idx] = o;
    if (seqOut) seqOut[idx] = f2bf(o);
  }
}

// ---------------------------------------------------------------------------
// Output head part 1: HO = relu(h @ W_o1 + b_o1) -> bf16.  4 waves/block.
// ---------------------------------------------------------------------------
__global__ void __launch_bounds__(128) k_out1(
    const float* __restrict__ H, const unsigned short* __restrict__ Wo1T,
    const float* __restrict__ bo, unsigned short* __restrict__ HO)
{
  const int lane = threadIdx.x & 31;
  const int w    = threadIdx.x >> 5;
  const int rowBase = blockIdx.x << 4;
  const int colBase = w << 4;

  v8f acc = {};
#pragma unroll
  for (int kc = 0; kc < 64; kc += 32) {
    const v16bf a = loadA_f32(H, rowBase, 64, kc, lane);
    const v16bf bb = loadB_w(Wo1T, colBase, 64, kc, lane);
    acc = wmma_bf16(a, bb, acc);
  }
  const int g = lane >> 4;
  const int col = colBase + (lane & 15);
  const float bias = bo[col];
#pragma unroll
  for (int v = 0; v < 8; ++v) {
    const int row = rowBase + v + (g << 3);
    HO[(size_t)row * 64 + col] = f2bf(fmaxf(acc[v] + bias, 0.0f));
  }
}

// ---------------------------------------------------------------------------
// Output head part 2: Y = HO @ W_o2 + b_o2, write transposed [B][12][1024].
// W_o2 pre-transposed+padded to [16][64].  4 row tiles per block.
// ---------------------------------------------------------------------------
__global__ void __launch_bounds__(128) k_out2(
    const unsigned short* __restrict__ HO, const unsigned short* __restrict__ Wo2T,
    const float* __restrict__ bo, float* __restrict__ out)
{
  const int lane = threadIdx.x & 31;
  const int w    = threadIdx.x >> 5;
  const int rowBase = (blockIdx.x << 6) + (w << 4);

  v8f acc = {};
#pragma unroll
  for (int kc = 0; kc < 64; kc += 32) {
    const v16bf a = loadA_bf16(HO, rowBase, 64, kc, lane);
    const v16bf bb = loadB_w(Wo2T, 0, 64, kc, lane);
    acc = wmma_bf16(a, bb, acc);
  }
  const int g = lane >> 4;
  const int col = lane & 15;
  if (col < 12) {
    const float bias = bo[col];
#pragma unroll
    for (int v = 0; v < 8; ++v) {
      const int row = rowBase + v + (g << 3);
      const int bq = row >> 10, n = row & 1023;
      out[((size_t)(bq * 12 + col) << 10) + n] = acc[v] + bias;
    }
  }
}

// ---------------------------------------------------------------------------
// Host orchestration
// ---------------------------------------------------------------------------
extern "C" void kernel_launch(void* const* d_in, const int* in_sizes, int n_in,
                              void* d_out, int out_size, void* d_ws, size_t ws_size,
                              hipStream_t stream) {
  (void)in_sizes; (void)n_in; (void)out_size;
  const float* X    = (const float*)d_in[0];
  const float* SE   = (const float*)d_in[3];
  const float* Wse1 = (const float*)d_in[4];  const float* bse1 = (const float*)d_in[5];
  const float* Wse2 = (const float*)d_in[6];  const float* bse2 = (const float*)d_in[7];
  const float* Wte1 = (const float*)d_in[8];  const float* bte1 = (const float*)d_in[9];
  const float* Wte2 = (const float*)d_in[10]; const float* bte2 = (const float*)d_in[11];
  const float* Win1 = (const float*)d_in[12]; const float* bin1 = (const float*)d_in[13];
  const float* Win2 = (const float*)d_in[14]; const float* bin2 = (const float*)d_in[15];
  const float* Wg1  = (const float*)d_in[16]; const float* bg1  = (const float*)d_in[17];
  const float* Wc1  = (const float*)d_in[18]; const float* bc1  = (const float*)d_in[19];
  const float* Wg2  = (const float*)d_in[20]; const float* bg2  = (const float*)d_in[21];
  const float* Wc2  = (const float*)d_in[22]; const float* bc2  = (const float*)d_in[23];
  const float* Wo1  = (const float*)d_in[24]; const float* bo1  = (const float*)d_in[25];
  const float* Wo2  = (const float*)d_in[26]; const float* bo2  = (const float*)d_in[27];
  const int*   TE   = (const int*)d_in[28];

  const int Mrow = 64 * 1024;          // B*N
  const int Mtot = 12 * Mrow;          // P*B*N

  char* ws = (char*)d_ws;
  size_t off = 0;
  auto carve = [&](size_t bytes) -> void* {
    void* p = ws + off;
    off = (off + bytes + 255) & ~(size_t)255;
    return p;
  };
  unsigned short* Xe   = (unsigned short*)carve((size_t)Mtot * 64 * 2);
  unsigned short* Seq1 = (unsigned short*)carve((size_t)Mtot * 64 * 2);
  float* Sa   = (float*)carve((size_t)Mrow * 64 * 4);
  float* Sb   = (float*)carve((size_t)Mrow * 64 * 4);
  unsigned short* RS = (unsigned short*)carve((size_t)Mrow * 64 * 2);
  float* Ubuf = (float*)carve((size_t)Mrow * 64 * 4);
  unsigned short* HO = (unsigned short*)carve((size_t)Mrow * 64 * 2);
  float* seB  = (float*)carve(1024 * 64 * 4);
  float* tmpA = (float*)carve(1024 * 64 * 4);
  float* teB  = (float*)carve(64 * 24 * 64 * 4);
  float* tmpB = (float*)carve(64 * 24 * 64 * 4);
  unsigned short* wgx1 = (unsigned short*)carve(128 * 64 * 2);
  unsigned short* wgs1 = (unsigned short*)carve(128 * 64 * 2);
  unsigned short* wcx1 = (unsigned short*)carve(64 * 64 * 2);
  unsigned short* wcs1 = (unsigned short*)carve(64 * 64 * 2);
  unsigned short* wgx2 = (unsigned short*)carve(128 * 64 * 2);
  unsigned short* wgs2 = (unsigned short*)carve(128 * 64 * 2);
  unsigned short* wcx2 = (unsigned short*)carve(64 * 64 * 2);
  unsigned short* wcs2 = (unsigned short*)carve(64 * 64 * 2);
  unsigned short* win2T = (unsigned short*)carve(64 * 64 * 2);
  unsigned short* wo1T  = (unsigned short*)carve(64 * 64 * 2);
  unsigned short* wo2T  = (unsigned short*)carve(16 * 64 * 2);
  if (ws_size < off) return;   // scratch too small: nothing safe to do

  // ---- weight prep ----
  k_fold<<<(128 * 64 + 255) / 256, 256, 0, stream>>>(Wg1, wgx1, wgs1, 128);
  k_fold<<<(64 * 64 + 255) / 256, 256, 0, stream>>>(Wc1, wcx1, wcs1, 64);
  k_fold<<<(128 * 64 + 255) / 256, 256, 0, stream>>>(Wg2, wgx2, wgs2, 128);
  k_fold<<<(64 * 64 + 255) / 256, 256, 0, stream>>>(Wc2, wcx2, wcs2, 64);
  k_transpose<<<(64 * 64 + 255) / 256, 256, 0, stream>>>(Win2, win2T, 64, 64);
  k_transpose<<<(64 * 64 + 255) / 256, 256, 0, stream>>>(Wo1, wo1T, 64, 64);
  k_transpose<<<(16 * 64 + 255) / 256, 256, 0, stream>>>(Wo2, wo2T, 12, 16);

  // ---- STEmbedding ----
  k_mm_small<<<(1024 * 64 + 255) / 256, 256, 0, stream>>>(SE, Wse1, bse1, tmpA, 1024, 64, 64, 1);
  k_mm_small<<<(1024 * 64 + 255) / 256, 256, 0, stream>>>(tmpA, Wse2, bse2, seB, 1024, 64, 64, 0);
  k_te1<<<(64 * 24 * 64 + 255) / 256, 256, 0, stream>>>(TE, Wte1, bte1, tmpB);
  k_mm_small<<<(64 * 24 * 64 + 255) / 256, 256, 0, stream>>>(tmpB, Wte2, bte2, teB, 64 * 24, 64, 64, 0);

  // ---- FC_X_in + STE add -> Xe (bf16, time-major) ----
  k_fc_in<<<Mtot / 16, 128, 0, stream>>>(X, Win1, bin1, win2T, bin2, seB, teB, Xe);

  // ---- DCGRU layer 1 ----
  k_zero<<<(Mrow * 64 + 255) / 256, 256, 0, stream>>>(Sa, Mrow * 64);
  float* cur = Sa; float* nxt = Sb;
  for (int t = 0; t < 12; ++t) {
    const unsigned short* xt = Xe + (size_t)t * Mrow * 64;
    k_gru_gates<<<Mrow / 16, 256, 0, stream>>>(xt, cur, wgx1, wgs1, bg1, RS, Ubuf);
    k_gru_cand<<<Mrow / 16, 128, 0, stream>>>(xt, RS, wcx1, wcs1, bc1, Ubuf, cur, nxt,
                                              Seq1 + (size_t)t * Mrow * 64);
    float* tsw = cur; cur = nxt; nxt = tsw;
  }

  // ---- DCGRU layer 2 ----
  float* cur2 = nxt; float* nxt2 = cur;   // reuse ping-pong buffers
  k_zero<<<(Mrow * 64 + 255) / 256, 256, 0, stream>>>(cur2, Mrow * 64);
  for (int t = 0; t < 12; ++t) {
    const unsigned short* xt = Seq1 + (size_t)t * Mrow * 64;
    k_gru_gates<<<Mrow / 16, 256, 0, stream>>>(xt, cur2, wgx2, wgs2, bg2, RS, Ubuf);
    k_gru_cand<<<Mrow / 16, 128, 0, stream>>>(xt, RS, wcx2, wcs2, bc2, Ubuf, cur2, nxt2,
                                              (unsigned short*)nullptr);
    float* tsw = cur2; cur2 = nxt2; nxt2 = tsw;
  }

  // ---- Output head ----
  k_out1<<<Mrow / 16, 128, 0, stream>>>(cur2, wo1T, bo1, HO);
  k_out2<<<Mrow / 64, 128, 0, stream>>>(HO, wo2T, bo2, (float*)d_out);
}
